// NeuralSDE_24292335026739
// MI455X (gfx1250) — compile-verified, matrix operand order checked by reading, not run on hardware
//
#include <hip/hip_runtime.h>

#define B_  4096
#define L_  128
#define I_  64
#define H_  256
#define D_  32
#define T_  101
#define G4H 1024   // 4*H
#define HD  288    // H + D
#define KC  320    // H + I : unified K for [h | x] @ [W_hh | W_ih]^T
#define MT  32     // batch rows per WG (2 x 16 M-tiles)

typedef __attribute__((ext_vector_type(16))) _Float16 v16h;
typedef __attribute__((ext_vector_type(8)))  float    v8f;
typedef __attribute__((ext_vector_type(2)))  float    v2f;

union FragH { v16h v; uint4 q[2]; _Float16 h[16]; };

__device__ __forceinline__ float sigf(float x) { return 1.0f / (1.0f + __expf(-x)); }
__device__ __forceinline__ float tanhfast(float x) {
    float e = __expf(2.0f * x);
    return (e - 1.0f) / (e + 1.0f);
}
__device__ __forceinline__ v8f zero8() {
    v8f z = {0.f, 0.f, 0.f, 0.f, 0.f, 0.f, 0.f, 0.f};
    return z;
}

// ---------------------------------------------------------------------------
// Kernel 0: pack Wcat[n][0:256] = W_hh[n][:], Wcat[n][256:320] = W_ih[n][:]
// as f16. Row stride 320 f16 = 640 B (16B-aligned): every B-fragment is two
// contiguous 16-byte chunks per lane.
// ---------------------------------------------------------------------------
__global__ void prep_weights(const float* __restrict__ Whh, const float* __restrict__ Wih,
                             _Float16* __restrict__ Wcat) {
    int i = blockIdx.x * blockDim.x + threadIdx.x;
    if (i >= G4H * KC) return;
    int n = i / KC, k = i % KC;
    float v = (k < H_) ? Whh[n * H_ + k] : Wih[n * I_ + (k - H_)];
    Wcat[i] = (_Float16)v;
}

// ---------------------------------------------------------------------------
// Kernel 1: LSTM encoder. 128 WGs x 256 threads (8 waves), 32-row batch tile
// per WG, all 128 steps private to the WG. Per step one [32,320]x[320,1024]
// GEMM: 10 K-tiles x 8 N-tiles x 2 M-tiles = 160 v_wmma_f32_16x16x32_f16 per
// wave, each streamed B-fragment reused for both M-tiles (32 FLOP/byte).
// Wave w owns gate columns [32w,32w+32) of i,f,g,o -> c/h update is pure
// register math; [h_t | x_t] double-buffered in 40 KB LDS, one barrier/step.
// ---------------------------------------------------------------------------
__global__ __launch_bounds__(256)
void lstm_kernel(const float* __restrict__ x,        // [B, L, I] f32
                 const float* __restrict__ b_ih,     // [4H]
                 const float* __restrict__ b_hh,     // [4H]
                 const _Float16* __restrict__ Wcat,  // [4H, 320] f16
                 float* __restrict__ h_out)          // [B, H] f32
{
    __shared__ _Float16 comb[2][MT][KC];             // 40 KB double buffer

    const int tid  = threadIdx.x;
    const int w    = tid >> 5;       // wave id 0..7
    const int l    = tid & 31;
    const int half = l >> 4;
    const int c16  = l & 15;
    const int bbase = blockIdx.x * MT;

    // prologue: h0 = 0, x_0 into buffer 0
    for (int i = tid; i < MT * H_; i += 256)
        comb[0][i >> 8][i & 255] = (_Float16)0.0f;
    for (int i = tid; i < MT * I_; i += 256) {
        int row = i >> 6, col = i & 63;
        comb[0][row][H_ + col] = (_Float16)x[(size_t)(bbase + row) * L_ * I_ + col];
    }
    __syncthreads();

    float c[2][2][8];                                 // [mi][s][r]
#pragma unroll
    for (int mi = 0; mi < 2; ++mi)
#pragma unroll
        for (int s = 0; s < 2; ++s)
#pragma unroll
            for (int r = 0; r < 8; ++r) c[mi][s][r] = 0.0f;

    // N-tile bases: j = region*2 + s, region in {i,f,g,o}
    int nb[8];
    float bias[8];
#pragma unroll
    for (int j = 0; j < 8; ++j) {
        int region = j >> 1, s = j & 1;
        int nbase = region * 256 + (2 * w + s) * 16;
        nb[j] = nbase;
        bias[j] = b_ih[nbase + c16] + b_hh[nbase + c16];
    }

    for (int t = 0; t < L_; ++t) {
        const int cur = t & 1;
        const int nxt = cur ^ 1;

        // Launder an OFFSET (not the pointer): defeats LICM on the 80
        // loop-invariant B-fragments (no 640-VGPR hoist + scratch spill)
        // while keeping global-pointer provenance -> global_load_b128,
        // not flat_load_b128.
        size_t woff = 0;
        asm volatile("" : "+s"(woff));
        const _Float16* Wt = Wcat + woff;

        v8f acc[2][8];
#pragma unroll
        for (int mi = 0; mi < 2; ++mi)
#pragma unroll
            for (int j = 0; j < 8; ++j) acc[mi][j] = zero8();

#pragma unroll
        for (int kt = 0; kt < 10; ++kt) {
            FragH a0, a1;
            const _Float16* hp0 = &comb[cur][c16][kt * 32 + half * 8];
            const _Float16* hp1 = &comb[cur][16 + c16][kt * 32 + half * 8];
            a0.q[0] = *(const uint4*)(hp0);
            a0.q[1] = *(const uint4*)(hp0 + 16);
            a1.q[0] = *(const uint4*)(hp1);
            a1.q[1] = *(const uint4*)(hp1 + 16);
#pragma unroll
            for (int j = 0; j < 8; ++j) {
                FragH b;
                const _Float16* wp = Wt + (size_t)(nb[j] + c16) * KC + kt * 32 + half * 8;
                b.q[0] = *(const uint4*)(wp);
                b.q[1] = *(const uint4*)(wp + 16);
                acc[0][j] = __builtin_amdgcn_wmma_f32_16x16x32_f16(
                    false, a0.v, false, b.v, (short)0, acc[0][j], false, false);
                acc[1][j] = __builtin_amdgcn_wmma_f32_16x16x32_f16(
                    false, a1.v, false, b.v, (short)0, acc[1][j], false, false);
            }
        }

        // gates + cell/hidden update (registers only), h_{t+1} -> comb[nxt]
#pragma unroll
        for (int mi = 0; mi < 2; ++mi) {
#pragma unroll
            for (int s = 0; s < 2; ++s) {
                const int hcol = (2 * w + s) * 16 + c16;
#pragma unroll
                for (int r = 0; r < 8; ++r) {
                    float gi = acc[mi][0 * 2 + s][r] + bias[0 * 2 + s];
                    float gf = acc[mi][1 * 2 + s][r] + bias[1 * 2 + s];
                    float gg = acc[mi][2 * 2 + s][r] + bias[2 * 2 + s];
                    float go = acc[mi][3 * 2 + s][r] + bias[3 * 2 + s];
                    float cc = sigf(gf) * c[mi][s][r] + sigf(gi) * tanhfast(gg);
                    float hh = sigf(go) * tanhfast(cc);
                    c[mi][s][r] = cc;
                    comb[nxt][mi * 16 + r + half * 8][hcol] = (_Float16)hh;
                    if (t == L_ - 1)
                        h_out[(size_t)(bbase + mi * 16 + r + half * 8) * H_ + hcol] = hh;
                }
            }
        }

        // x_{t+1} -> comb[nxt] (cooperative, coalesced)
        if (t + 1 < L_) {
            for (int i = tid; i < MT * I_; i += 256) {
                int row = i >> 6, col = i & 63;
                comb[nxt][row][H_ + col] =
                    (_Float16)x[(size_t)(bbase + row) * L_ * I_ + (t + 1) * I_ + col];
            }
        }
        __syncthreads();
    }
}

// ---------------------------------------------------------------------------
// Kernel 2: Euler-Maruyama SDE. 256 WGs x 32 threads (one wave, 16-row batch
// tile). Prologue hoists the loop-invariant hid @ W[:, :H].T projections
// (v_wmma_f32_16x16x4_f32, K=256), then 100 steps of y @ W[:, H:].T plus the
// elementwise update; state stays in f32 registers. Here register-hoisting of
// the 32 small B-fragments across the loop is beneficial, so no laundering.
// ---------------------------------------------------------------------------
__global__ __launch_bounds__(32)
void sde_kernel(const float* __restrict__ ts,        // [T]
                const float* __restrict__ y0,        // [B, D]
                const float* __restrict__ dWn,       // [T-1, B, D]
                const float* __restrict__ driftW,    // [D, H+D]
                const float* __restrict__ driftB,    // [D]
                const float* __restrict__ diffW,     // [D, H+D]
                const float* __restrict__ diffB,     // [D]
                const float* __restrict__ hsrc,      // [B, H]
                float* __restrict__ out)             // [T, B, D]
{
    __shared__ float ylds[16][D_];

    const int l    = threadIdx.x;
    const int half = l >> 4;
    const int c16  = l & 15;
    const int bbase = blockIdx.x * 16;

    // ---- loop-invariant: cacc[q] = hid @ W[:, :H].T + bias  (q = mat*2+nt) ----
    v8f cacc[4];
#pragma unroll
    for (int q = 0; q < 4; ++q) cacc[q] = zero8();

    const float* hrow = hsrc + (size_t)(bbase + c16) * H_;
    for (int tk = 0; tk < 64; ++tk) {
        v2f a = *(const v2f*)(hrow + tk * 4 + half * 2);
#pragma unroll
        for (int mat = 0; mat < 2; ++mat) {
            const float* W = mat ? diffW : driftW;
#pragma unroll
            for (int nt = 0; nt < 2; ++nt) {
                v2f b = *(const v2f*)(W + (size_t)(nt * 16 + c16) * HD + tk * 4 + half * 2);
                cacc[mat * 2 + nt] = __builtin_amdgcn_wmma_f32_16x16x4_f32(
                    false, a, false, b, (short)0, cacc[mat * 2 + nt], false, false);
            }
        }
    }
#pragma unroll
    for (int nt = 0; nt < 2; ++nt) {
        float bd = driftB[nt * 16 + c16];
        float bg = diffB[nt * 16 + c16];
#pragma unroll
        for (int r = 0; r < 8; ++r) {
            cacc[0 + nt][r] += bd;
            cacc[2 + nt][r] += bg;
        }
    }

    // ---- y = y0 (C layout), emit out[0] ----
    v8f y[2];
#pragma unroll
    for (int nt = 0; nt < 2; ++nt)
#pragma unroll
        for (int r = 0; r < 8; ++r) {
            int m = r + half * 8;
            float v = y0[(size_t)(bbase + m) * D_ + nt * 16 + c16];
            y[nt][r] = v;
            out[(size_t)(bbase + m) * D_ + nt * 16 + c16] = v;
        }

    for (int t = 0; t < T_ - 1; ++t) {
        // stage y into LDS so A-fragments can be gathered in WMMA layout
#pragma unroll
        for (int nt = 0; nt < 2; ++nt)
#pragma unroll
            for (int r = 0; r < 8; ++r)
                ylds[r + half * 8][nt * 16 + c16] = y[nt][r];
        __syncthreads();

        v8f facc[4];
#pragma unroll
        for (int q = 0; q < 4; ++q) facc[q] = zero8();

#pragma unroll
        for (int tk = 0; tk < 8; ++tk) {
            v2f a = *(const v2f*)(&ylds[c16][tk * 4 + half * 2]);
#pragma unroll
            for (int mat = 0; mat < 2; ++mat) {
                const float* W = mat ? diffW : driftW;
#pragma unroll
                for (int nt = 0; nt < 2; ++nt) {
                    v2f b = *(const v2f*)(W + (size_t)(nt * 16 + c16) * HD + H_ + tk * 4 + half * 2);
                    facc[mat * 2 + nt] = __builtin_amdgcn_wmma_f32_16x16x4_f32(
                        false, a, false, b, (short)0, facc[mat * 2 + nt], false, false);
                }
            }
        }

        float dt = ts[t + 1] - ts[t];
        float sq = sqrtf(dt);

#pragma unroll
        for (int nt = 0; nt < 2; ++nt)
#pragma unroll
            for (int r = 0; r < 8; ++r) {
                int m = r + half * 8;
                float dw = dWn[((size_t)t * B_ + bbase + m) * D_ + nt * 16 + c16];
                float f = cacc[0 + nt][r] + facc[0 + nt][r];
                float g = cacc[2 + nt][r] + facc[2 + nt][r];
                float yn = y[nt][r] + f * dt + g * (dw * sq);
                y[nt][r] = yn;
                out[((size_t)(t + 1) * B_ + bbase + m) * D_ + nt * 16 + c16] = yn;
            }
        __syncthreads();
    }
}

// ---------------------------------------------------------------------------
extern "C" void kernel_launch(void* const* d_in, const int* in_sizes, int n_in,
                              void* d_out, int out_size, void* d_ws, size_t ws_size,
                              hipStream_t stream) {
    const float* ts   = (const float*)d_in[0];
    const float* y0   = (const float*)d_in[1];
    const float* xseq = (const float*)d_in[2];
    const float* dWn  = (const float*)d_in[3];
    const float* W_ih = (const float*)d_in[4];
    const float* W_hh = (const float*)d_in[5];
    const float* b_ih = (const float*)d_in[6];
    const float* b_hh = (const float*)d_in[7];
    const float* drW  = (const float*)d_in[8];
    const float* drB  = (const float*)d_in[9];
    const float* dfW  = (const float*)d_in[10];
    const float* dfB  = (const float*)d_in[11];
    float* out = (float*)d_out;

    char* ws = (char*)d_ws;
    _Float16* Wcat = (_Float16*)ws;                    // 1024*320*2 = 640 KB
    float*    hF32 = (float*)(ws + 1024 * 1024);       // 4 MB

    prep_weights<<<(G4H * KC + 255) / 256, 256, 0, stream>>>(W_hh, W_ih, Wcat);
    lstm_kernel<<<B_ / MT, 256, 0, stream>>>(xseq, b_ih, b_hh, Wcat, hF32);
    sde_kernel<<<B_ / 16, 32, 0, stream>>>(ts, y0, dWn, drW, drB, dfW, dfB, hF32, out);
}
